// SambaYAttention_89043261981032
// MI455X (gfx1250) — matrix operand order repaired and checked by reading, not compiled, not run on hardware
//
#include <hip/hip_runtime.h>
#include <math.h>

// ---------------------------------------------------------------------------
// CDNA5 (gfx1250) wave32 WMMA types / helpers
// ---------------------------------------------------------------------------
typedef __attribute__((ext_vector_type(16))) __bf16 v16bf;
typedef __attribute__((ext_vector_type(8)))  float  v8f;

union Frag { v16bf bf; uint4 q[2]; };

__device__ __forceinline__ unsigned short f2bf(float f) {
  union { float f; unsigned u; } v; v.f = f;
  unsigned r = v.u + 0x7FFFu + ((v.u >> 16) & 1u);   // round-to-nearest-even
  return (unsigned short)(r >> 16);
}

// Low 32 bits of a generic pointer to __shared__ = LDS byte offset (ISA 10.2).
__device__ __forceinline__ unsigned lds_off(const void* p) {
  return (unsigned)(uintptr_t)p;
}
// Async copy 16B/lane global -> LDS, tracked by ASYNCcnt (ISA 10.x / 15.18.3).
__device__ __forceinline__ void a2l_b128(const unsigned short* g, unsigned l) {
  asm volatile("global_load_async_to_lds_b128 %0, %1, off"
               :: "v"(l), "v"(g) : "memory");
}
__device__ __forceinline__ void wait_async4() {
  asm volatile("s_wait_asynccnt 0x4" ::: "memory");
}
__device__ __forceinline__ void wait_async0() {
  asm volatile("s_wait_asynccnt 0x0" ::: "memory");
}

// A operand (16x32 bf16, ISA 7.12.2): lane<16 -> row=lane, K={kb..kb+7, kb+16..kb+23}
//                                     lane>=16 -> row=lane-16, K={kb+8..kb+15, kb+24..kb+31}
__device__ __forceinline__ v16bf load_fragA(const unsigned short* base, int row0,
                                            int ldk, int kbase, int lane) {
  int hi = lane >> 4, r = lane & 15;
  const unsigned short* p = base + (size_t)(row0 + r) * ldk + kbase + (hi ? 8 : 0);
  Frag f; f.q[0] = *(const uint4*)p; f.q[1] = *(const uint4*)(p + 16);
  return f.bf;
}
// B operand (32x16 bf16): lane<16 -> N=lane, K={kb..kb+7, kb+16..kb+23} ; lane>=16 -> N=lane-16, K={kb+8..kb+15, kb+24..kb+31}
__device__ __forceinline__ v16bf load_fragB(const unsigned short* base, int row0,
                                            int ldk, int kbase, int lane) {
  int hi = lane >> 4, r = lane & 15;
  const unsigned short* p = base + (size_t)(row0 + r) * ldk + kbase + (hi ? 8 : 0);
  Frag f; f.q[0] = *(const uint4*)p; f.q[1] = *(const uint4*)(p + 16);
  return f.bf;
}

__device__ __forceinline__ v8f wmma_bf16(v16bf a, v16bf b, v8f c) {
  return __builtin_amdgcn_wmma_f32_16x16x32_bf16(false, a, false, b, (short)0, c,
                                                 false, false);
}

#define TT   2048
#define OPW  6144
#define LAMBDA_INIT 0.7455691397f   // 0.8 - 0.6*exp(-0.3*8)

// ---------------------------------------------------------------------------
// fp32 -> bf16 pre-conversion (8 elems / thread, exact-size grids)
// ---------------------------------------------------------------------------
__global__ __launch_bounds__(256) void cvt_bf16_kernel(
    const float* __restrict__ s, unsigned short* __restrict__ d) {
  size_t i = ((size_t)blockIdx.x * 256 + threadIdx.x) * 8;
  float4 f0 = *(const float4*)(s + i);
  float4 f1 = *(const float4*)(s + i + 4);
  union { unsigned short u[8]; uint4 q; } o;
  o.u[0] = f2bf(f0.x); o.u[1] = f2bf(f0.y); o.u[2] = f2bf(f0.z); o.u[3] = f2bf(f0.w);
  o.u[4] = f2bf(f1.x); o.u[5] = f2bf(f1.y); o.u[6] = f2bf(f1.z); o.u[7] = f2bf(f1.w);
  *(uint4*)(d + i) = o.q;
}

// ---------------------------------------------------------------------------
// bf16 WMMA GEMM:  C[M,N] = A[M,K] * Bw[N,K]^T + bias
// 128x128 tile, BK=32, double-buffered async global->LDS staging.
// 8 waves each own a 32x64 sub-tile.
// ---------------------------------------------------------------------------
template<int OBF16>
__global__ __launch_bounds__(256) void gemm_wmma_kernel(
    const unsigned short* __restrict__ A, const unsigned short* __restrict__ Bw,
    const float* __restrict__ bias, void* __restrict__ Cptr,
    int M, int N, int K) {
  __shared__ __attribute__((aligned(16))) unsigned short As[2][128 * 40];
  __shared__ __attribute__((aligned(16))) unsigned short Bs[2][128 * 40];
  const int tid = threadIdx.x, lane = tid & 31, wid = tid >> 5;
  const int m0 = blockIdx.y * 128, n0 = blockIdx.x * 128;
  const int wm = (wid >> 1) * 32;
  const int wn = (wid & 1) * 64;
  v8f acc[2][4] = {};

  const int lrow = tid >> 1, lch = (tid & 1) * 16;
  const unsigned short* agp = A  + (size_t)(m0 + lrow) * K + lch;
  const unsigned short* bgp = Bw + (size_t)(n0 + lrow) * K + lch;
  unsigned aoff[2], boff[2];
  aoff[0] = lds_off(&As[0][lrow * 40 + lch]);
  aoff[1] = lds_off(&As[1][lrow * 40 + lch]);
  boff[0] = lds_off(&Bs[0][lrow * 40 + lch]);
  boff[1] = lds_off(&Bs[1][lrow * 40 + lch]);

  auto issue = [&](int kb, int b) {
    a2l_b128(agp + kb,     aoff[b]);
    a2l_b128(agp + kb + 8, aoff[b] + 16);
    a2l_b128(bgp + kb,     boff[b]);
    a2l_b128(bgp + kb + 8, boff[b] + 16);
  };

  const int NK = K >> 5;
  issue(0, 0);
  for (int i = 0; i < NK; ++i) {
    const int cur = i & 1;
    if (i + 1 < NK) { issue((i + 1) << 5, cur ^ 1); wait_async4(); }
    else            { wait_async0(); }
    __syncthreads();
    v16bf a0 = load_fragA(As[cur], wm,      40, 0, lane);
    v16bf a1 = load_fragA(As[cur], wm + 16, 40, 0, lane);
#pragma unroll
    for (int j = 0; j < 4; ++j) {
      v16bf b = load_fragB(Bs[cur], wn + j * 16, 40, 0, lane);
      acc[0][j] = wmma_bf16(a0, b, acc[0][j]);
      acc[1][j] = wmma_bf16(a1, b, acc[1][j]);
    }
    __syncthreads();
  }

  const int hi = lane >> 4, ln = lane & 15;
#pragma unroll
  for (int i = 0; i < 2; ++i)
#pragma unroll
    for (int j = 0; j < 4; ++j)
#pragma unroll
      for (int r = 0; r < 8; ++r) {
        int row = m0 + wm + i * 16 + r + hi * 8;
        int col = n0 + wn + j * 16 + ln;
        float v = acc[i][j][r] + bias[col];
        if (OBF16) ((unsigned short*)Cptr)[(size_t)row * N + col] = f2bf(v);
        else       ((float*)Cptr)[(size_t)row * N + col] = v;
      }
}

// ---------------------------------------------------------------------------
// Flash attention per (head-pair, component, 64-row q block).
// Q frags in registers; K tile via async global->LDS; V^T staged via VGPRs;
// S via WMMA -> LDS -> online softmax -> P (bf16) -> WMMA P*V with rescale.
// ---------------------------------------------------------------------------
__global__ __launch_bounds__(256) void attn_flash_kernel(
    const unsigned short* __restrict__ qkv, float* __restrict__ Ows) {
  __shared__ __attribute__((aligned(16))) unsigned short Ks[64 * 136];
  __shared__ __attribute__((aligned(16))) unsigned short Vt[256 * 72];
  __shared__ __attribute__((aligned(16))) float          Sb[64 * 68];
  __shared__ __attribute__((aligned(16))) unsigned short Pb[64 * 72];
  __shared__ float corr_s[64];
  __shared__ float linv_s[64];

  const int tid = threadIdx.x, lane = tid & 31, wid = tid >> 5;
  const int qb = blockIdx.x, p = blockIdx.y, c = blockIdx.z;
  const int kvp  = p >> 2;                       // GQA: 4 query pairs per kv pair
  const int qoff = (2 * p + c) * 128;
  const int koff = 4096 + (2 * kvp + c) * 128;
  const int voff = 4096 + 2048 + kvp * 256;
  const float scale = 0.08838834764831845f;      // 1/sqrt(128)

  const int mtile = (wid >> 1) * 16;
  const int ebase = (wid & 1) * 128;
  const int hi = lane >> 4, ln = lane & 15;

  v16bf qa[4];
#pragma unroll
  for (int kk = 0; kk < 4; ++kk) {
    int t = qb * 64 + mtile + ln;
    const unsigned short* pq = qkv + (size_t)t * OPW + qoff + kk * 32 + (hi ? 8 : 0);
    Frag f; f.q[0] = *(const uint4*)pq; f.q[1] = *(const uint4*)(pq + 16);
    qa[kk] = f.bf;
  }

  // per-thread staging addresses
  const int krow = tid >> 2, kcb = (tid & 3) * 32;
  const unsigned kdst = lds_off(Ks + krow * 136 + kcb);

  v8f o[8] = {};
  float m_i = -__builtin_inff(), l_i = 0.f;      // live only in threads 0..63

  for (int kb = 0; kb <= qb; ++kb) {
    __syncthreads();
    { // K tile via async copy (4 x b128 per lane)
      const unsigned short* src = qkv + (size_t)(kb * 64 + krow) * OPW + koff + kcb;
      a2l_b128(src,      kdst);
      a2l_b128(src + 8,  kdst + 16);
      a2l_b128(src + 16, kdst + 32);
      a2l_b128(src + 24, kdst + 48);
    }
    { // V tile, transposed into Vt[e][k] (layout change -> through VGPRs)
      int k = tid >> 2, eb = (tid & 3) * 64;
      const unsigned short* src = qkv + (size_t)(kb * 64 + k) * OPW + voff + eb;
      uint4 u[8];
#pragma unroll
      for (int j = 0; j < 8; ++j) u[j] = *(const uint4*)(src + j * 8);
      const unsigned short* us = (const unsigned short*)u;
#pragma unroll
      for (int j = 0; j < 64; ++j) Vt[(eb + j) * 72 + k] = us[j];
    }
    wait_async0();
    __syncthreads();
    { // S = Q K^T : wave computes two 16x16 tiles at cols n0, n0+16
      int n0 = (wid & 1) * 32;
      v8f s0 = {}, s1 = {};
#pragma unroll
      for (int kk = 0; kk < 4; ++kk) {
        v16bf b0 = load_fragB(Ks, n0,      136, kk * 32, lane);
        v16bf b1 = load_fragB(Ks, n0 + 16, 136, kk * 32, lane);
        s0 = wmma_bf16(qa[kk], b0, s0);
        s1 = wmma_bf16(qa[kk], b1, s1);
      }
#pragma unroll
      for (int r = 0; r < 8; ++r) {
        Sb[(mtile + r + hi * 8) * 68 + n0 + ln]      = s0[r];
        Sb[(mtile + r + hi * 8) * 68 + n0 + 16 + ln] = s1[r];
      }
    }
    __syncthreads();
    if (tid < 64) { // online softmax, one thread per query row
      int m = tid;
      int nvalid = (kb < qb) ? 64 : (m + 1);
      const float* row = Sb + m * 68;
      float mx = m_i;
      for (int n = 0; n < nvalid; ++n) mx = fmaxf(mx, row[n] * scale);
      float corr = __expf(m_i - mx);
      float sum = 0.f;
      for (int n = 0; n < 64; ++n) {
        float pv = 0.f;
        if (n < nvalid) { pv = __expf(row[n] * scale - mx); sum += pv; }
        Pb[m * 72 + n] = f2bf(pv);
      }
      l_i = l_i * corr + sum;
      m_i = mx;
      corr_s[m] = corr;
    }
    __syncthreads();
    { // rescale running O, then O += P * V
      float cf[8];
#pragma unroll
      for (int r = 0; r < 8; ++r) cf[r] = corr_s[mtile + r + hi * 8];
#pragma unroll
      for (int j = 0; j < 8; ++j)
#pragma unroll
        for (int r = 0; r < 8; ++r) o[j][r] *= cf[r];
#pragma unroll
      for (int kk = 0; kk < 2; ++kk) {
        v16bf pa = load_fragA(Pb, mtile, 72, kk * 32, lane);
#pragma unroll
        for (int j = 0; j < 8; ++j) {
          v16bf vb = load_fragB(Vt, ebase + j * 16, 72, kk * 32, lane);
          o[j] = wmma_bf16(pa, vb, o[j]);
        }
      }
    }
  }
  __syncthreads();
  if (tid < 64) linv_s[tid] = 1.0f / l_i;
  __syncthreads();
  {
    float lf[8];
#pragma unroll
    for (int r = 0; r < 8; ++r) lf[r] = linv_s[mtile + r + hi * 8];
    size_t base = ((size_t)(c * 16 + p) * TT + qb * 64) * 256;
#pragma unroll
    for (int j = 0; j < 8; ++j)
#pragma unroll
      for (int r = 0; r < 8; ++r)
        Ows[base + (size_t)(mtile + r + hi * 8) * 256 + ebase + j * 16 + ln] =
            o[j][r] * lf[r];
  }
}

// ---------------------------------------------------------------------------
// Differential combine + per-head-pair RMSNorm, emits bf16 for final GEMM.
// ---------------------------------------------------------------------------
__global__ __launch_bounds__(256) void combine_kernel(
    const float* __restrict__ Ows,
    const float* __restrict__ lq1, const float* __restrict__ lk1,
    const float* __restrict__ lq2, const float* __restrict__ lk2,
    const float* __restrict__ subw, unsigned short* __restrict__ attn_bf) {
  __shared__ float red[256];
  __shared__ float lam_s;
  const int t = blockIdx.x, p = blockIdx.y, e = threadIdx.x;
  if (e == 0) {
    float s1 = 0.f, s2 = 0.f;
    for (int i = 0; i < 128; ++i) { s1 += lq1[i] * lk1[i]; s2 += lq2[i] * lk2[i]; }
    lam_s = __expf(s1) - __expf(s2) + LAMBDA_INIT;
  }
  __syncthreads();
  size_t i0 = ((size_t)p * TT + t) * 256 + e;
  size_t i1 = ((size_t)(16 + p) * TT + t) * 256 + e;
  float x = Ows[i0] - lam_s * Ows[i1];
  red[e] = x * x;
  __syncthreads();
  for (int s = 128; s > 0; s >>= 1) {
    if (e < s) red[e] += red[e + s];
    __syncthreads();
  }
  float rms = rsqrtf(red[0] * (1.0f / 256.0f) + 1e-5f);
  float y = x * rms * subw[e] * (1.0f - LAMBDA_INIT);
  attn_bf[(size_t)t * 4096 + p * 256 + e] = f2bf(y);
}

// ---------------------------------------------------------------------------
extern "C" void kernel_launch(void* const* d_in, const int* in_sizes, int n_in,
                              void* d_out, int out_size, void* d_ws, size_t ws_size,
                              hipStream_t stream) {
  (void)in_sizes; (void)n_in; (void)out_size; (void)ws_size;
  const float* hidden = (const float*)d_in[0];
  const float* wqkv_w = (const float*)d_in[1];
  const float* wqkv_b = (const float*)d_in[2];
  const float* lq1    = (const float*)d_in[3];
  const float* lk1    = (const float*)d_in[4];
  const float* lq2    = (const float*)d_in[5];
  const float* lk2    = (const float*)d_in[6];
  const float* subw   = (const float*)d_in[7];
  const float* out_w  = (const float*)d_in[8];
  const float* out_b  = (const float*)d_in[9];

  char* ws = (char*)d_ws;
  unsigned short* qkv_ws = (unsigned short*)ws;                    // 24 MiB
  float*          O_ws   = (float*)(ws + 25165824);                // 64 MiB
  unsigned short* hid_bf = (unsigned short*)(ws + 92274688);       // 16 MiB (reused as attn_bf)
  unsigned short* attn_bf = hid_bf;   // lifetime-disjoint alias
  unsigned short* wqkv_bf = (unsigned short*)(ws + 109051904);     // 48 MiB
  unsigned short* outw_bf = (unsigned short*)(ws + 159383552);     // 32 MiB

  // 0) one-time bf16 conversions (weights stay resident in L2 across M-blocks)
  cvt_bf16_kernel<<<4096,  256, 0, stream>>>(hidden, hid_bf);      // 2048*4096
  cvt_bf16_kernel<<<12288, 256, 0, stream>>>(wqkv_w, wqkv_bf);     // 6144*4096
  cvt_bf16_kernel<<<8192,  256, 0, stream>>>(out_w,  outw_bf);     // 4096*4096
  // 1) QKV projection -> bf16 (+bias)
  gemm_wmma_kernel<1><<<dim3(48, 16), 256, 0, stream>>>(
      hid_bf, wqkv_bf, wqkv_b, qkv_ws, 2048, 6144, 4096);
  // 2) differential flash attention, both softmax components
  attn_flash_kernel<<<dim3(32, 16, 2), 256, 0, stream>>>(qkv_ws, O_ws);
  // 3) combine + RMSNorm -> bf16 activations
  combine_kernel<<<dim3(2048, 16), 256, 0, stream>>>(
      O_ws, lq1, lk1, lq2, lk2, subw, attn_bf);
  // 4) output projection -> fp32 (+bias)
  gemm_wmma_kernel<0><<<dim3(32, 16), 256, 0, stream>>>(
      attn_bf, outw_bf, out_b, d_out, 2048, 4096, 4096);
}
